// FusedColorMLP_46471546142970
// MI455X (gfx1250) — matrix operand order repaired and checked by reading, not compiled, not run on hardware
//
#include <hip/hip_runtime.h>

typedef __attribute__((ext_vector_type(16))) _Float16 v16h;
typedef __attribute__((ext_vector_type(8)))  _Float16 v8h;
typedef __attribute__((ext_vector_type(8)))  float    v8f;
typedef __attribute__((ext_vector_type(4)))  float    v4f;

constexpr int IN_DIM  = 155;
constexpr int KPAD    = 160;   // 5 chunks of 32
constexpr int HIDDEN  = 128;
constexpr int OUT_DIM = 3;

static __device__ inline v16h cat16(v8h lo, v8h hi) {
  return __builtin_shufflevector(lo, hi, 0,1,2,3,4,5,6,7,8,9,10,11,12,13,14,15);
}

// LDS layout (in halves):
//   [0, 128*160)                      W1s : B-operand layout, [n][k] column-contiguous K (zero padded)
//   [128*160, 128*160+16*128)         W2s : [c=16][k=128], cols 3..15 zero
//   then 8 waves * (16*160)           Xs  : per-wave x tile (f16); reused as Hs (16*128) for GEMM2
__global__ void __launch_bounds__(256)
fused_color_mlp(const float* __restrict__ X,
                const float* __restrict__ W1,
                const float* __restrict__ W2,
                float* __restrict__ Out,
                int nrows)
{
  extern __shared__ _Float16 smem[];
  _Float16* W1s = smem;
  _Float16* W2s = smem + 128 * KPAD;

  const int tid  = threadIdx.x;
  const int lane = tid & 31;
  const int wave = tid >> 5;
  _Float16* Xs = smem + 128 * KPAD + 16 * HIDDEN + wave * (16 * KPAD);

  // ---- stage W1 transposed into f16 LDS: coalesced b128 global reads,
  //      scattered b16 LDS writes. W1 is [k=155][n=128] row-major; 128%4==0
  //      so each float4 stays within one k-row. ----
  for (int s4 = tid; s4 < (IN_DIM * HIDDEN) / 4; s4 += 256) {
    v4f v = *(const v4f*)(W1 + s4 * 4);
    int k = (s4 * 4) / HIDDEN;
    int n = (s4 * 4) - k * HIDDEN;
    #pragma unroll
    for (int j = 0; j < 4; ++j)
      W1s[(n + j) * KPAD + k] = (_Float16)v[j];
  }
  // zero-pad K rows 155..159 across all 128 columns
  for (int i = tid; i < 128 * (KPAD - IN_DIM); i += 256) {
    int n = i / (KPAD - IN_DIM);
    int k = IN_DIM + (i - n * (KPAD - IN_DIM));
    W1s[n * KPAD + k] = (_Float16)0.0f;
  }
  // ---- stage W2 padded to 16 output columns (tiny) ----
  for (int idx = tid; idx < 16 * HIDDEN; idx += 256) {
    int c = idx >> 7;
    int k = idx & 127;
    float v = (c < OUT_DIM) ? W2[k * OUT_DIM + c] : 0.0f;
    W2s[idx] = (_Float16)v;
  }
  __syncthreads();

  const int rowBase = blockIdx.x * 128 + wave * 16;

  // ---- stage this wave's 16x160 x tile as f16 ----
  if (rowBase + 16 <= nrows) {
    // Fast path: the 16 rows are one contiguous 16B-aligned span of
    // 16*155 = 2480 floats = 620 float4s -> coalesced global_load_b128.
    const float* xt = X + (long)rowBase * IN_DIM;
    for (int i = lane; i < (16 * IN_DIM) / 4; i += 32) {
      v4f v = *(const v4f*)(xt + i * 4);
      #pragma unroll
      for (int j = 0; j < 4; ++j) {
        int e = i * 4 + j;
        int r = e / IN_DIM;
        int k = e - r * IN_DIM;
        Xs[r * KPAD + k] = (_Float16)v[j];
      }
    }
    // zero the 5 pad columns of each row (16*5 = 80 halves)
    for (int i = lane; i < 16 * (KPAD - IN_DIM); i += 32) {
      int r = i / (KPAD - IN_DIM);
      int k = IN_DIM + (i - r * (KPAD - IN_DIM));
      Xs[r * KPAD + k] = (_Float16)0.0f;
    }
  } else {
    // Tail path (unused when nrows % 128 == 0): clamped scalar loads.
    for (int i = lane; i < 16 * KPAD; i += 32) {
      int r = i / KPAD;
      int k = i - r * KPAD;
      int row = rowBase + r;
      if (row >= nrows) row = nrows - 1;
      float v = (k < IN_DIM) ? X[(long)row * IN_DIM + k] : 0.0f;
      Xs[i] = (_Float16)v;
    }
  }

  const int m  = lane & 15;
  const int hi = (lane >> 4) & 1;

  // ================= GEMM1: h[16,128] = x[16,160] @ W1[160,128] =================
  v8f acc[8];
  #pragma unroll
  for (int t = 0; t < 8; ++t) acc[t] = v8f{};

  #pragma unroll
  for (int kc = 0; kc < 5; ++kc) {
    const int k0 = kc * 32;
    // A fragment: lane<16 -> K {k0+0..7, k0+16..23}; lane>=16 -> K {k0+8..15, k0+24..31}
    const _Float16* ap = Xs + m * KPAD + k0 + (hi ? 8 : 0);
    v8h alo = *(const v8h*)ap;
    v8h ahi = *(const v8h*)(ap + 16);
    v16h a = cat16(alo, ahi);
    #pragma unroll
    for (int nt = 0; nt < 8; ++nt) {
      // B fragment: column n = nt*16+m; lane<16 -> K {k0+0..7, k0+16..23}; lane>=16 -> K {k0+8..15, k0+24..31}
      const _Float16* bp = W1s + (nt * 16 + m) * KPAD + k0 + (hi ? 8 : 0);
      v8h blo = *(const v8h*)bp;
      v8h bhi = *(const v8h*)(bp + 16);
      v16h b = cat16(blo, bhi);
      acc[nt] = __builtin_amdgcn_wmma_f32_16x16x32_f16(
          false, a, false, b, (short)0, acc[nt], false, false);
    }
  }

  // ---- ReLU and spill h (f16) into the reused per-wave LDS region ----
  _Float16* Hs = Xs;   // 16*128 <= 16*160, x tile fully consumed
  #pragma unroll
  for (int nt = 0; nt < 8; ++nt) {
    #pragma unroll
    for (int g = 0; g < 8; ++g) {
      float v = acc[nt][g];
      v = v > 0.0f ? v : 0.0f;
      int M = g + (hi << 3);                 // C/D layout: VGPR g, lane half
      Hs[M * HIDDEN + nt * 16 + m] = (_Float16)v;
    }
  }

  // ================= GEMM2: out[16,16] = h[16,128] @ W2p[128,16] =================
  v8f acc2 = v8f{};
  #pragma unroll
  for (int kc = 0; kc < 4; ++kc) {
    const int k0 = kc * 32;
    const _Float16* ap = Hs + m * HIDDEN + k0 + (hi ? 8 : 0);
    v8h alo = *(const v8h*)ap;
    v8h ahi = *(const v8h*)(ap + 16);
    v16h a = cat16(alo, ahi);
    const _Float16* bp = W2s + m * HIDDEN + k0 + (hi ? 8 : 0);
    v8h blo = *(const v8h*)bp;
    v8h bhi = *(const v8h*)(bp + 16);
    v16h b = cat16(blo, bhi);
    acc2 = __builtin_amdgcn_wmma_f32_16x16x32_f16(
        false, a, false, b, (short)0, acc2, false, false);
  }

  // ---- store the 3 real output columns ----
  if (m < OUT_DIM) {
    #pragma unroll
    for (int g = 0; g < 8; ++g) {
      int row = rowBase + g + (hi << 3);
      if (row < nrows) Out[(long)row * OUT_DIM + m] = acc2[g];
    }
  }
}

extern "C" void kernel_launch(void* const* d_in, const int* in_sizes, int n_in,
                              void* d_out, int out_size, void* d_ws, size_t ws_size,
                              hipStream_t stream) {
  const float* X  = (const float*)d_in[0];
  const float* W1 = (const float*)d_in[1];
  const float* W2 = (const float*)d_in[2];
  float* Out = (float*)d_out;

  int nrows = in_sizes[0] / IN_DIM;            // 1048576
  int grid  = (nrows + 127) / 128;             // 128 rows per block (8 waves x 16 rows)
  size_t ldsBytes = (size_t)(128 * KPAD + 16 * HIDDEN + 8 * 16 * KPAD) * 2;  // 86016 B

  fused_color_mlp<<<grid, 256, ldsBytes, stream>>>(X, W1, W2, Out, nrows);
}
